// MultiHeadAttention_46162308498209
// MI455X (gfx1250) — matrix operand order repaired and checked by reading, not compiled, not run on hardware
//
#include <hip/hip_runtime.h>

// ---------------- problem constants ----------------
#define HEADS   8
#define DKH     128
#define DMODEL  1024
#define BATCH   8
#define SEQ     1024
#define MTOT    (BATCH*SEQ)

typedef __attribute__((ext_vector_type(16))) __bf16 v16bf;
typedef __attribute__((ext_vector_type(8)))  __bf16 v8bf;
typedef __attribute__((ext_vector_type(8)))  float  v8f;
typedef __attribute__((ext_vector_type(4)))  float  v4f;
typedef __attribute__((ext_vector_type(8)))  unsigned v8u;
typedef __attribute__((ext_vector_type(4)))  unsigned v4u;
typedef int v4i __attribute__((vector_size(16)));   // matches async-builtin pointee

#if __has_builtin(__builtin_amdgcn_global_load_async_to_lds_b128) && \
    __has_builtin(__builtin_amdgcn_s_wait_asynccnt)
#define USE_ASYNC 1
#else
#define USE_ASYNC 0
#endif

// f32 -> bf16 round-to-nearest-even (scalar)
__device__ __forceinline__ __bf16 f2bf(float f) {
  unsigned u = __builtin_bit_cast(unsigned, f);
  unsigned r = (u + 0x7FFFu + ((u >> 16) & 1u)) >> 16;
  unsigned short s = (unsigned short)r;
  return __builtin_bit_cast(__bf16, s);
}

// two f32 -> packed bf16x2 in a uint (RNE both)
__device__ __forceinline__ unsigned pk_bf16(float a, float b) {
  unsigned ua = __builtin_bit_cast(unsigned, a);
  unsigned ub = __builtin_bit_cast(unsigned, b);
  ua = (ua + 0x7FFFu + ((ua >> 16) & 1u)) >> 16;
  ub = (ub + 0x7FFFu + ((ub >> 16) & 1u));
  return (ua & 0xFFFFu) | (ub & 0xFFFF0000u);
}

// A-matrix 16x32 bf16 fragment (ISA 7.12.2): lane<16 holds K 0..7,16..23 ;
// lane>=16 holds K 8..15,24..31.
__device__ __forceinline__ v16bf frag_a(const __bf16* p, int ld, int row, int k0, int lane) {
  const __bf16* r = p + row * ld + k0 + ((lane & 16) ? 8 : 0);
  v8bf lo = *(const v8bf*)(r);
  v8bf hi = *(const v8bf*)(r + 16);
  v16bf f;
#pragma unroll
  for (int i = 0; i < 8; ++i) { f[i] = lo[i]; f[i + 8] = hi[i]; }
  return f;
}

// B-matrix 32x16 bf16 fragment: lane(n)<16 holds K 0..7,16..23 ; lane>=16 holds K 8..15,24..31.
__device__ __forceinline__ v16bf frag_b(const __bf16* p, int ld, int row, int k0, int lane) {
  const __bf16* r = p + row * ld + k0 + ((lane & 16) ? 8 : 0);
  v8bf lo = *(const v8bf*)(r);
  v8bf hi = *(const v8bf*)(r + 16);
  v16bf f;
#pragma unroll
  for (int i = 0; i < 8; ++i) { f[i] = lo[i]; f[i + 8] = hi[i]; }
  return f;
}

#define WMMA_BF16(a, b, c) \
  __builtin_amdgcn_wmma_f32_16x16x32_bf16(false, (a), false, (b), (short)0, (c), false, false)

#if USE_ASYNC
// 16-byte global -> LDS async copy (cache -> LDS, no VGPR round trip, ASYNCcnt)
#define ASYNC_CP16(gsrc, ldst)                                              \
  __builtin_amdgcn_global_load_async_to_lds_b128(                           \
      (__attribute__((address_space(1))) v4i*)(gsrc),                       \
      (__attribute__((address_space(3))) v4i*)(ldst), 0, 0)
#endif

// ---------------- weight f32 -> bf16 convert ----------------
__global__ __launch_bounds__(256) void cvt_kernel(const float* __restrict__ src,
                                                  __bf16* __restrict__ dst, int n) {
  int i = (blockIdx.x * 256 + threadIdx.x) * 8;
  if (i + 7 < n) {
    v4f x0 = *(const v4f*)(src + i);
    v4f x1 = *(const v4f*)(src + i + 4);
    v4u o;
    o[0] = pk_bf16(x0[0], x0[1]); o[1] = pk_bf16(x0[2], x0[3]);
    o[2] = pk_bf16(x1[0], x1[1]); o[3] = pk_bf16(x1[2], x1[3]);
    *(v4u*)(dst + i) = o;
  }
}

// ---------------- projection: out = (X @ W^T + b) * alpha ----------------
// Software-pipelined: next A chunk loads issued behind current WMMA chain;
// four B fragments live in distinct registers for partial loadcnt waits.
// vtrans==0: out[(b*H+h)*SEQ + s][dk] ; vtrans==1: out[(b*H+h)*DKH + dk][s]
__global__ __launch_bounds__(256) void proj_kernel(const float* __restrict__ X,
                                                   const __bf16* __restrict__ W,
                                                   const float* __restrict__ bias,
                                                   __bf16* __restrict__ out,
                                                   float alpha, int vtrans) {
  const int m_base = blockIdx.x * 128, n_base = blockIdx.y * 64;
  const int wave = threadIdx.x >> 5, lane = threadIdx.x & 31, l16 = lane & 15;
  v8f acc[4] = {};

  // per-lane A row pointer with K-split offset baked in
  const float* ar = X + (size_t)(m_base + wave * 16 + l16) * DMODEL + ((lane & 16) ? 8 : 0);

  v4f x0 = *(const v4f*)(ar);
  v4f x1 = *(const v4f*)(ar + 4);
  v4f x2 = *(const v4f*)(ar + 16);
  v4f x3 = *(const v4f*)(ar + 20);

  for (int kc = 0; kc < DMODEL; kc += 32) {
    v8u au;
    au[0] = pk_bf16(x0[0], x0[1]); au[1] = pk_bf16(x0[2], x0[3]);
    au[2] = pk_bf16(x1[0], x1[1]); au[3] = pk_bf16(x1[2], x1[3]);
    au[4] = pk_bf16(x2[0], x2[1]); au[5] = pk_bf16(x2[2], x2[3]);
    au[6] = pk_bf16(x3[0], x3[1]); au[7] = pk_bf16(x3[2], x3[3]);
    v16bf a = __builtin_bit_cast(v16bf, au);

    // issue next A-chunk loads now (clamped: last iteration refetches, harmless)
    const int kn = (kc + 32 < DMODEL) ? kc + 32 : kc;
    x0 = *(const v4f*)(ar + kn);
    x1 = *(const v4f*)(ar + kn + 4);
    x2 = *(const v4f*)(ar + kn + 16);
    x3 = *(const v4f*)(ar + kn + 20);

    v16bf b0 = frag_b(W, DMODEL, n_base + l16,      kc, lane);
    v16bf b1 = frag_b(W, DMODEL, n_base + 16 + l16, kc, lane);
    v16bf b2 = frag_b(W, DMODEL, n_base + 32 + l16, kc, lane);
    v16bf b3 = frag_b(W, DMODEL, n_base + 48 + l16, kc, lane);
    acc[0] = WMMA_BF16(a, b0, acc[0]);
    acc[1] = WMMA_BF16(a, b1, acc[1]);
    acc[2] = WMMA_BF16(a, b2, acc[2]);
    acc[3] = WMMA_BF16(a, b3, acc[3]);
  }

  // epilogue: bias, scale, scatter-store bf16
  const int rbase = m_base + wave * 16 + ((lane & 16) ? 8 : 0);
#pragma unroll
  for (int sub = 0; sub < 4; ++sub) {
    int n = n_base + sub * 16 + l16;
    float bv = bias[n];
    int h = n >> 7, dk = n & 127;
#pragma unroll
    for (int j = 0; j < 8; ++j) {
      int mrow = rbase + j;
      int b = mrow >> 10, s = mrow & 1023;
      float val = (acc[sub][j] + bv) * alpha;
      size_t idx = vtrans ? (((size_t)(b * HEADS + h) * DKH + dk) * SEQ + s)
                          : (((size_t)(b * HEADS + h) * SEQ + s) * DKH + dk);
      out[idx] = f2bf(val);
    }
  }
}

// ---------------- flash attention (per b,h ; 128 q rows per block) ----------------
// Q,K: bf16 [B*H, SEQ, DKH] (Q pre-scaled by 1/sqrt(DKH)); VT: bf16 [B*H, DKH, SEQ]
// CTX out: bf16 [B*SEQ, DMODEL] laid out [b, s, h*DKH + dk]
__global__ __launch_bounds__(256) void attn_kernel(const __bf16* __restrict__ Q,
                                                   const __bf16* __restrict__ K,
                                                   const __bf16* __restrict__ VT,
                                                   __bf16* __restrict__ CTX) {
  __shared__ __bf16 kT[32 * 136];       // K chunk: 32 kcols x 128 d, ld=136
  __shared__ __bf16 vT[128 * 40];       // V^T chunk: 128 d x 32 kcols, ld=40
  __shared__ __bf16 pT[8 * 16 * 40];    // per-wave P tile: 16 x 32, ld=40

  const int bh = blockIdx.x;
  const int q_base = blockIdx.y * 128;
  const int wave = threadIdx.x >> 5, lane = threadIdx.x & 31, l16 = lane & 15;

  // resident Q fragments for this wave's 16 rows (whole DK=128, 4 K-chunks)
  const __bf16* qp = Q + ((size_t)bh * SEQ + q_base + wave * 16 + l16) * DKH;
  v16bf qf[4];
#pragma unroll
  for (int dc = 0; dc < 4; ++dc) qf[dc] = frag_a(qp, 0, 0, dc * 32, lane);

  v8f ctx[8] = {};
  float m[8], l[8];
#pragma unroll
  for (int j = 0; j < 8; ++j) { m[j] = -1e30f; l[j] = 0.f; }

  const __bf16* Kb = K + (size_t)bh * SEQ * DKH;
  const __bf16* Vb = VT + (size_t)bh * DKH * SEQ;
  const int sr8 = threadIdx.x >> 3, sc8 = (threadIdx.x & 7) * 16;
  const int sr2 = threadIdx.x >> 1, sc2 = (threadIdx.x & 1) * 16;

  for (int kc = 0; kc < SEQ; kc += 32) {
    __syncthreads();
    const __bf16* srcK = Kb + (size_t)(kc + sr8) * DKH + sc8;
    const __bf16* srcV = Vb + (size_t)sr2 * SEQ + kc + sc2;
    __bf16* dstK = kT + sr8 * 136 + sc8;
    __bf16* dstV = vT + sr2 * 40 + sc2;
#if USE_ASYNC
    ASYNC_CP16(srcK,     dstK);
    ASYNC_CP16(srcK + 8, dstK + 8);
    ASYNC_CP16(srcV,     dstV);
    ASYNC_CP16(srcV + 8, dstV + 8);
    if (kc + 32 < SEQ) {  // prefetch next chunk while async copies run
      __builtin_prefetch(srcK + 32 * DKH, 0, 0);
      __builtin_prefetch(srcV + 32, 0, 0);
    }
    __builtin_amdgcn_s_wait_asynccnt(0);
#else
    *(v8bf*)(dstK)     = *(const v8bf*)(srcK);
    *(v8bf*)(dstK + 8) = *(const v8bf*)(srcK + 8);
    *(v8bf*)(dstV)     = *(const v8bf*)(srcV);
    *(v8bf*)(dstV + 8) = *(const v8bf*)(srcV + 8);
#endif
    __syncthreads();

    // scores 16x32 (two 16x16 N-subtiles), K = DK = 128
    v8f s0 = {}, s1 = {};
#pragma unroll
    for (int dc = 0; dc < 4; ++dc) {
      v16bf b0 = frag_b(kT, 136, l16,      dc * 32, lane);
      v16bf b1 = frag_b(kT, 136, 16 + l16, dc * 32, lane);
      s0 = WMMA_BF16(qf[dc], b0, s0);
      s1 = WMMA_BF16(qf[dc], b1, s1);
    }

    // online softmax update (row r = j + (lane>=16?8:0); 16 cols per half-wave)
    float corr[8];
#pragma unroll
    for (int j = 0; j < 8; ++j) {
      float rm = fmaxf(s0[j], s1[j]);
      rm = fmaxf(rm, __shfl_xor(rm, 1, 32));
      rm = fmaxf(rm, __shfl_xor(rm, 2, 32));
      rm = fmaxf(rm, __shfl_xor(rm, 4, 32));
      rm = fmaxf(rm, __shfl_xor(rm, 8, 32));
      float mn = fmaxf(m[j], rm);
      corr[j] = __expf(m[j] - mn);
      m[j] = mn;
      float p0 = __expf(s0[j] - mn);
      float p1 = __expf(s1[j] - mn);
      s0[j] = p0; s1[j] = p1;
      float rs = p0 + p1;
      rs += __shfl_xor(rs, 1, 32);
      rs += __shfl_xor(rs, 2, 32);
      rs += __shfl_xor(rs, 4, 32);
      rs += __shfl_xor(rs, 8, 32);
      l[j] = l[j] * corr[j] + rs;
    }
#pragma unroll
    for (int ns = 0; ns < 8; ++ns)
#pragma unroll
      for (int j = 0; j < 8; ++j) ctx[ns][j] *= corr[j];

    // C-layout -> A-layout via per-wave LDS tile
    __bf16* pw = pT + wave * 16 * 40;
    const int rofs = (lane & 16) ? 8 : 0;
#pragma unroll
    for (int j = 0; j < 8; ++j) {
      pw[(j + rofs) * 40 + l16]      = f2bf(s0[j]);
      pw[(j + rofs) * 40 + l16 + 16] = f2bf(s1[j]);
    }
    __syncthreads();

    v16bf pa = frag_a(pw, 40, l16, 0, lane);
#pragma unroll
    for (int ns = 0; ns < 8; ++ns) {
      v16bf vb = frag_b(vT, 40, ns * 16 + l16, 0, lane);
      ctx[ns] = WMMA_BF16(pa, vb, ctx[ns]);
    }
  }

  // normalize and store ctx as bf16 [b, s, h*DKH + dk]
  float inv[8];
#pragma unroll
  for (int j = 0; j < 8; ++j) inv[j] = 1.0f / l[j];
  const int b = bh >> 3, h = bh & 7;
  const int rbase = q_base + wave * 16 + ((lane & 16) ? 8 : 0);
#pragma unroll
  for (int ns = 0; ns < 8; ++ns) {
    int col = h * DKH + ns * 16 + l16;
#pragma unroll
    for (int j = 0; j < 8; ++j) {
      int srow = rbase + j;
      CTX[((size_t)(b * SEQ + srow)) * DMODEL + col] = f2bf(ctx[ns][j] * inv[j]);
    }
  }
}

// ---------------- arrange + output projection: out = packed(ctx) @ Wo^T + bo ----------------
// Arrange is fused into the per-lane A-fragment gather (no LDS, no barriers).
__global__ __launch_bounds__(256) void out_kernel(const __bf16* __restrict__ CTX,
                                                  const __bf16* __restrict__ Wo,
                                                  const float* __restrict__ bo,
                                                  const int* __restrict__ cfg,
                                                  float* __restrict__ out) {
  const int m_base = blockIdx.x * 128, n_base = blockIdx.y * 64;
  const int wave = threadIdx.x >> 5, lane = threadIdx.x & 31, l16 = lane & 15;
  v8f acc[4] = {};

  const int mrow = m_base + wave * 16 + l16;       // this lane's A row
  const int d = 32 * (cfg[mrow >> 10] + 1);        // D_LIST = 32*(i+1); uniform per block
  const int kmax = 8 * d;
  const int k0 = (lane & 16) ? 8 : 0;
  const __bf16* ctxr = CTX + (size_t)mrow * DMODEL;
  const __bf16 zero = f2bf(0.0f);

  for (int kc = 0; kc < DMODEL; kc += 32) {
    v16bf a;
#pragma unroll
    for (int i = 0; i < 8; ++i) {
      int kA = kc + k0 + i;        // elements 0..7
      int kB = kA + 16;            // elements 8..15
      __bf16 va = zero, vb = zero;
      if (kA < kmax) { int hh = kA / d; va = ctxr[hh * DKH + (kA - hh * d)]; }
      if (kB < kmax) { int hh = kB / d; vb = ctxr[hh * DKH + (kB - hh * d)]; }
      a[i] = va; a[i + 8] = vb;
    }
    v16bf b0 = frag_b(Wo, DMODEL, n_base + l16,      kc, lane);
    v16bf b1 = frag_b(Wo, DMODEL, n_base + 16 + l16, kc, lane);
    v16bf b2 = frag_b(Wo, DMODEL, n_base + 32 + l16, kc, lane);
    v16bf b3 = frag_b(Wo, DMODEL, n_base + 48 + l16, kc, lane);
    acc[0] = WMMA_BF16(a, b0, acc[0]);
    acc[1] = WMMA_BF16(a, b1, acc[1]);
    acc[2] = WMMA_BF16(a, b2, acc[2]);
    acc[3] = WMMA_BF16(a, b3, acc[3]);
  }

  const int rbase = m_base + wave * 16 + ((lane & 16) ? 8 : 0);
#pragma unroll
  for (int sub = 0; sub < 4; ++sub) {
    int n = n_base + sub * 16 + l16;
    float bv = bo[n];
#pragma unroll
    for (int j = 0; j < 8; ++j)
      out[(size_t)(rbase + j) * DMODEL + n] = acc[sub][j] + bv;
  }
}

// ---------------- launcher ----------------
extern "C" void kernel_launch(void* const* d_in, const int* in_sizes, int n_in,
                              void* d_out, int out_size, void* d_ws, size_t ws_size,
                              hipStream_t stream) {
  (void)in_sizes; (void)n_in; (void)out_size;
  const float* query = (const float*)d_in[0];
  const float* key   = (const float*)d_in[1];
  const float* value = (const float*)d_in[2];
  const int*   cfg   = (const int*)d_in[3];
  const float* Wq = (const float*)d_in[4];  const float* bq = (const float*)d_in[5];
  const float* Wk = (const float*)d_in[6];  const float* bk = (const float*)d_in[7];
  const float* Wv = (const float*)d_in[8];  const float* bv = (const float*)d_in[9];
  const float* Wo = (const float*)d_in[10]; const float* bo = (const float*)d_in[11];

  char* ws = (char*)d_ws;
  const size_t WW = (size_t)DMODEL * DMODEL * sizeof(__bf16);            // 2 MB per weight
  const size_t PT = (size_t)BATCH * HEADS * SEQ * DKH * sizeof(__bf16);  // 16 MB
  const size_t need = 4 * WW + 4 * PT;                                   // 72 MB
  if (ws_size < need) return;

  __bf16* Wq_b = (__bf16*)(ws);
  __bf16* Wk_b = (__bf16*)(ws + WW);
  __bf16* Wv_b = (__bf16*)(ws + 2 * WW);
  __bf16* Wo_b = (__bf16*)(ws + 3 * WW);
  __bf16* Qw   = (__bf16*)(ws + 4 * WW);
  __bf16* Kw   = (__bf16*)(ws + 4 * WW + PT);
  __bf16* Vt   = (__bf16*)(ws + 4 * WW + 2 * PT);
  __bf16* Cw   = (__bf16*)(ws + 4 * WW + 3 * PT);

  const int nW = DMODEL * DMODEL;
  cvt_kernel<<<nW / 2048, 256, 0, stream>>>(Wq, Wq_b, nW);
  cvt_kernel<<<nW / 2048, 256, 0, stream>>>(Wk, Wk_b, nW);
  cvt_kernel<<<nW / 2048, 256, 0, stream>>>(Wv, Wv_b, nW);
  cvt_kernel<<<nW / 2048, 256, 0, stream>>>(Wo, Wo_b, nW);

  dim3 gp(MTOT / 128, DMODEL / 64);
  const float alpha = 0.08838834764831845f;  // 1/sqrt(128)
  proj_kernel<<<gp, 256, 0, stream>>>(query, Wq_b, bq, Qw, alpha, 0);
  proj_kernel<<<gp, 256, 0, stream>>>(key,   Wk_b, bk, Kw, 1.0f,  0);
  proj_kernel<<<gp, 256, 0, stream>>>(value, Wv_b, bv, Vt, 1.0f,  1);

  dim3 ga(BATCH * HEADS, SEQ / 128);
  attn_kernel<<<ga, 256, 0, stream>>>(Qw, Kw, Vt, Cw);

  dim3 go(MTOT / 128, DMODEL / 64);
  out_kernel<<<go, 256, 0, stream>>>(Cw, Wo_b, bo, cfg, (float*)d_out);
}
